// GraphChlorophyllNet_10703058502004
// MI455X (gfx1250) — matrix operand order repaired and checked by reading, not compiled, not run on hardware
//
#include <hip/hip_runtime.h>
#include <math.h>

#define NN 20000
#define TT 12
#define FF 9
#define HH 128
#define EE 320000
#define LHH 128
#define G4 512
#define D2 256

typedef __bf16 bf16;
typedef __attribute__((ext_vector_type(16))) __bf16 v16bf;
typedef __attribute__((ext_vector_type(8)))  float  v8f;
typedef __attribute__((ext_vector_type(4)))  unsigned u4;
typedef __attribute__((ext_vector_type(4)))  unsigned u32x4;
typedef __attribute__((ext_vector_type(8)))  int      i32x8;
typedef __attribute__((ext_vector_type(4)))  int      i32x4;

// ---------- helpers ----------
__device__ __forceinline__ bf16 to_bf16(float f) {
    unsigned u = __builtin_bit_cast(unsigned, f);
    unsigned r = u + 0x7FFFu + ((u >> 16) & 1u);   // round-to-nearest-even
    unsigned short h = (unsigned short)(r >> 16);
    return __builtin_bit_cast(bf16, h);
}
__device__ __forceinline__ float from_bf16(bf16 b) {
    unsigned short h = __builtin_bit_cast(unsigned short, b);
    unsigned u = ((unsigned)h) << 16;
    return __builtin_bit_cast(float, u);
}
__device__ __forceinline__ float wave_sum(float v) {
#pragma unroll
    for (int m = 16; m >= 1; m >>= 1) v += __shfl_xor(v, m, 32);
    return v;
}
__device__ __forceinline__ float sigm(float x) { return 1.f / (1.f + expf(-x)); }
__device__ __forceinline__ float gelu_exact(float x) {
    return 0.5f * x * (1.f + erff(x * 0.70710678118654752f));
}

// ---------- WMMA GEMM with TDM->LDS staging ----------
// C[M,Nc] = A[M,K](bf16 row-major) x B(pre-packed fragments) (+bias | +=C)
// block = 256 threads = 8 waves. Block tile: 64 rows x (up to 8*16 cols).
// Wave w -> col tile blockIdx.x*8+w, 4 row tiles of 16.
// A tile 64x32 bf16 staged into LDS by the Tensor Data Mover each K-step.
__global__ void wmma_gemm_kernel(const bf16* __restrict__ A, const bf16* __restrict__ Bp,
                                 float* __restrict__ C, const float* __restrict__ bias,
                                 int M, int K, int Nc, int accumulate) {
    __shared__ bf16 atile[64 * 32];   // 4 KB
    int wave = threadIdx.x >> 5;
    int lane = threadIdx.x & 31;
    int nct  = Nc >> 4;
    int ct   = blockIdx.x * 8 + wave;
    int m0   = blockIdx.y * 64;
    int half = lane >> 4, l15 = lane & 15;
    bool active = ct < nct;
    int bcol = ct * 16 + l15;

    v8f acc[4];
#pragma unroll
    for (int rt = 0; rt < 4; ++rt) acc[rt] = (v8f){};
    if (accumulate && active) {
#pragma unroll
        for (int rt = 0; rt < 4; ++rt) {
            int rb = m0 + rt * 16;
            if (rb < M) {
#pragma unroll
                for (int r = 0; r < 8; ++r)
                    acc[rt][r] = C[(size_t)(rb + r + half * 8) * Nc + bcol];
            }
        }
    }

    unsigned long long abase = (unsigned long long)(const void*)A +
                               (unsigned long long)m0 * (unsigned)K * 2ull;
    unsigned ldsoff = (unsigned)(unsigned long long)(void*)&atile[0];
    int nkb = K >> 5;

    for (int kb = 0; kb < nkb; ++kb) {
        __syncthreads();                       // LDS free to overwrite
        if (threadIdx.x < 32) {                // wave 0 drives the TDM
            unsigned long long ga = abase + ((unsigned long long)kb << 6); // +kb*32 elem *2B
            u32x4 g0;
            g0[0] = 1u;                                          // count=1, user mode
            g0[1] = ldsoff;                                      // LDS byte address
            g0[2] = (unsigned)ga;                                // global addr lo
            g0[3] = ((unsigned)(ga >> 32) & 0x01FFFFFFu) | 0x80000000u; // addr hi | type=2
            i32x8 g1;
            g1[0] = 0x00010000;                                  // data_size = 2 bytes
            g1[1] = (int)(((unsigned)K & 0xFFFFu) << 16);        // tensor_dim0 lo16
            g1[2] = (int)(((unsigned)K >> 16) | (((unsigned)M & 0xFFFFu) << 16)); // dim0 hi | dim1 lo
            g1[3] = (int)(((unsigned)M >> 16) | (32u << 16));    // dim1 hi | tile_dim0=32
            g1[4] = 64;                                          // tile_dim1=64, tile_dim2=0
            g1[5] = K;                                           // tensor_dim0_stride lo32
            g1[6] = 0;                                           // stride hi | dim1_stride lo
            g1[7] = 0;
            i32x4 z4 = (i32x4){0, 0, 0, 0};
            i32x8 z8 = (i32x8){0, 0, 0, 0, 0, 0, 0, 0};
            __builtin_amdgcn_tensor_load_to_lds(g0, g1, z4, z4, z8, 0);
            __builtin_amdgcn_s_wait_tensorcnt(0);
        }
        __syncthreads();                       // tile visible to all waves
        if (active) {
            if (kb + 1 < nkb)
                __builtin_prefetch((const void*)(Bp + (((size_t)(kb + 1) * nct + ct) << 9)), 0, 1);
            union { u4 q[2]; v16bf v; } bfr;
            const u4* bp = (const u4*)(Bp + (((size_t)kb * nct + ct) << 9)) + lane * 2;
            bfr.q[0] = bp[0];
            bfr.q[1] = bp[1];
#pragma unroll
            for (int rt = 0; rt < 4; ++rt) {
                union { u4 q[2]; v16bf v; } afr;
                const u4* rp = (const u4*)(atile + (rt * 16 + l15) * 32);
                afr.q[0] = rp[half * 2];
                afr.q[1] = rp[half * 2 + 1];
                acc[rt] = __builtin_amdgcn_wmma_f32_16x16x32_bf16(
                    false, afr.v, false, bfr.v, (short)0, acc[rt], false, false);
            }
        }
    }

    if (active) {
        float bv = bias ? bias[bcol] : 0.f;
#pragma unroll
        for (int rt = 0; rt < 4; ++rt) {
            int rb = m0 + rt * 16;
            if (rb < M) {
#pragma unroll
                for (int r = 0; r < 8; ++r)
                    C[(size_t)(rb + r + half * 8) * Nc + bcol] = acc[rt][r] + bv;
            }
        }
    }
}

// ---------- B-fragment pre-pack ----------
// dst chunk (kb, ct): 32 lanes x 16 bf16, lane holds column ct*16+(lane&15),
// K rows kb*32 + (lane>>4)*16 + j.  srcTransposed: src is [Nc,K] (Wih/Whh).
__global__ void pack_b_kernel(const float* __restrict__ src, bf16* __restrict__ dst,
                              int K, int Nc, int srcTransposed) {
    long idx = (long)blockIdx.x * blockDim.x + threadIdx.x;
    if (idx >= (long)K * Nc) return;
    int j = (int)(idx & 15);
    int lane = (int)((idx >> 4) & 31);
    long chunk = idx >> 9;
    int nct = Nc >> 4;
    int c = (int)(chunk % nct), kb = (int)(chunk / nct);
    int k = kb * 32 + (lane >> 4) * 16 + j;
    int n = c * 16 + (lane & 15);
    float v = srcTransposed ? src[(size_t)n * K + k] : src[(size_t)k * Nc + n];
    dst[idx] = to_bf16(v);
}

// ---------- misc small kernels ----------
__global__ void zero_u32_kernel(unsigned* p, long n) {
    long i = (long)blockIdx.x * blockDim.x + threadIdx.x;
    if (i < n) p[i] = 0u;
}

// ---------- degree / GCN ----------
__global__ void deg_init_kernel(float* deg) {
    int n = blockIdx.x * blockDim.x + threadIdx.x;
    if (n < NN) deg[n] = 1.f;                         // self-loop
}
__global__ void deg_accum_kernel(const int* ei, float* deg) {
    int e = blockIdx.x * blockDim.x + threadIdx.x;
    if (e < EE) atomicAdd(&deg[ei[EE + e]], 1.f);
}
__global__ void deg_finalize_kernel(float* di) {
    int n = blockIdx.x * blockDim.x + threadIdx.x;
    if (n < NN) di[n] = rsqrtf(di[n]);
}
// wave-per-node: y = GELU(LN(x_t @ in_W + b))
__global__ void input_proj_kernel(const float* __restrict__ x, const float* __restrict__ W,
                                  const float* __restrict__ b, const float* __restrict__ g,
                                  const float* __restrict__ beta, float* __restrict__ h,
                                  bf16* __restrict__ h_bf, int t) {
    int n = blockIdx.x * 8 + (threadIdx.x >> 5);
    int lane = threadIdx.x & 31;
    if (n >= NN) return;
    float xv[FF];
#pragma unroll
    for (int f = 0; f < FF; ++f) xv[f] = x[((size_t)n * TT + t) * FF + f];
    float y[4];
#pragma unroll
    for (int q = 0; q < 4; ++q) {
        int j = lane + 32 * q;
        float a = b[j];
#pragma unroll
        for (int f = 0; f < FF; ++f) a += xv[f] * W[f * HH + j];
        y[q] = a;
    }
    float mu = wave_sum(y[0] + y[1] + y[2] + y[3]) * (1.f / HH);
    float vv = 0.f;
#pragma unroll
    for (int q = 0; q < 4; ++q) { float d = y[q] - mu; vv += d * d; }
    float rstd = rsqrtf(wave_sum(vv) * (1.f / HH) + 1e-5f);
#pragma unroll
    for (int q = 0; q < 4; ++q) {
        int j = lane + 32 * q;
        float z = (y[q] - mu) * rstd * g[j] + beta[j];
        float ge = gelu_exact(z);
        h[(size_t)n * HH + j] = ge;
        h_bf[(size_t)n * HH + j] = to_bf16(ge);
    }
}
__global__ void agg_init_kernel(const float* m, const float* di, const float* gb, float* agg) {
    long i = (long)blockIdx.x * blockDim.x + threadIdx.x;
    if (i < (long)NN * HH) {
        int n = (int)(i >> 7), j = (int)(i & 127);
        float d = di[n];
        agg[i] = m[i] * d * d + gb[j];
    }
}
// one wave per edge; lane covers 4 features
__global__ void edge_scatter_kernel(const int* __restrict__ ei, const float* __restrict__ m,
                                    const float* __restrict__ di, float* __restrict__ agg) {
    long tid = (long)blockIdx.x * blockDim.x + threadIdx.x;
    if (tid >= (long)EE * 32) return;
    int e = (int)(tid >> 5), lane = (int)(tid & 31);
    int s = ei[e], d = ei[EE + e];
    float ne = di[s] * di[d];
    const float4 mv = *(const float4*)(m + (size_t)s * HH + lane * 4);
    float* ap = agg + (size_t)d * HH + lane * 4;
    atomicAdd(ap + 0, mv.x * ne);
    atomicAdd(ap + 1, mv.y * ne);
    atomicAdd(ap + 2, mv.z * ne);
    atomicAdd(ap + 3, mv.w * ne);
}
// wave-per-node: h = ReLU(LN(agg + h))
__global__ void gcn_ln_relu_kernel(const float* __restrict__ agg, const float* __restrict__ g,
                                   const float* __restrict__ beta, float* __restrict__ h,
                                   bf16* __restrict__ out_bf) {
    int n = blockIdx.x * 8 + (threadIdx.x >> 5);
    int lane = threadIdx.x & 31;
    if (n >= NN) return;
    float y[4];
#pragma unroll
    for (int q = 0; q < 4; ++q) {
        int j = lane + 32 * q;
        y[q] = agg[(size_t)n * HH + j] + h[(size_t)n * HH + j];
    }
    float mu = wave_sum(y[0] + y[1] + y[2] + y[3]) * (1.f / HH);
    float vv = 0.f;
#pragma unroll
    for (int q = 0; q < 4; ++q) { float d = y[q] - mu; vv += d * d; }
    float rstd = rsqrtf(wave_sum(vv) * (1.f / HH) + 1e-5f);
#pragma unroll
    for (int q = 0; q < 4; ++q) {
        int j = lane + 32 * q;
        float z = (y[q] - mu) * rstd * g[j] + beta[j];
        float r = fmaxf(z, 0.f);
        h[(size_t)n * HH + j] = r;
        out_bf[(size_t)n * HH + j] = to_bf16(r);
    }
}

// ---------- LSTM cell (pointwise) ----------
__global__ void lstm_cell_kernel(const float* __restrict__ gates, float* __restrict__ c,
                                 bf16* __restrict__ h_bf, bf16* __restrict__ out) {
    long idx = (long)blockIdx.x * blockDim.x + threadIdx.x;
    if (idx >= (long)NN * LHH) return;
    int n = (int)(idx >> 7), j = (int)(idx & 127);
    const float* gp = gates + (size_t)n * G4;
    float ig = sigm(gp[j]);
    float fg = sigm(gp[LHH + j]);
    float gg = tanhf(gp[2 * LHH + j]);
    float og = sigm(gp[3 * LHH + j]);
    float cn = fg * c[idx] + ig * gg;
    c[idx] = cn;
    float hn = og * tanhf(cn);
    bf16 hb = to_bf16(hn);
    h_bf[idx] = hb;
    out[(size_t)n * D2 + j] = hb;
}

// ---------- attention ----------
__global__ void attn_score_kernel(const float* __restrict__ ah, const float* __restrict__ W2,
                                  float* __restrict__ scores, int rows) {
    int r = blockIdx.x * blockDim.x + threadIdx.x;
    if (r >= rows) return;
    float s = 0.f;
    for (int j = 0; j < LHH; ++j) s += tanhf(ah[(size_t)r * LHH + j]) * W2[j];
    scores[r] = s;
}
__global__ void softmax_t_kernel(const float* __restrict__ scores, float* __restrict__ w) {
    int n = blockIdx.x * blockDim.x + threadIdx.x;
    if (n >= NN) return;
    float mx = -1e30f;
    for (int t = 0; t < TT; ++t) mx = fmaxf(mx, scores[(size_t)t * NN + n]);
    float e[TT]; float sum = 0.f;
    for (int t = 0; t < TT; ++t) { e[t] = expf(scores[(size_t)t * NN + n] - mx); sum += e[t]; }
    float inv = 1.f / sum;
    for (int t = 0; t < TT; ++t) w[(size_t)t * NN + n] = e[t] * inv;
}
__global__ void context_kernel(const float* __restrict__ w, const bf16* __restrict__ lo,
                               bf16* __restrict__ final_bf) {
    long idx = (long)blockIdx.x * blockDim.x + threadIdx.x;
    if (idx >= (long)NN * D2) return;
    int n = (int)(idx >> 8), cc = (int)(idx & 255);
    float acc = 0.f;
    for (int t = 0; t < TT; ++t)
        acc += w[(size_t)t * NN + n] * from_bf16(lo[((size_t)t * NN + n) * D2 + cc]);
    acc += from_bf16(lo[((size_t)(TT - 1) * NN + n) * D2 + cc]);
    final_bf[idx] = to_bf16(acc);
}

// ---------- output head ----------
__global__ void gelu_kernel(const float* in, float* out_f32, bf16* out_bf, long n) {
    long i = (long)blockIdx.x * blockDim.x + threadIdx.x;
    if (i >= n) return;
    float g = gelu_exact(in[i]);
    if (out_f32) out_f32[i] = g;
    if (out_bf) out_bf[i] = to_bf16(g);
}
__global__ void out3_kernel(const float* __restrict__ h2, const float* __restrict__ W3,
                            const float* __restrict__ b3, float* __restrict__ out) {
    int n = blockIdx.x * blockDim.x + threadIdx.x;
    if (n >= NN) return;
    for (int o = 0; o < 6; ++o) {
        float a = b3[o];
        for (int k = 0; k < 64; ++k) a += h2[(size_t)n * 64 + k] * W3[k * 6 + o];
        out[(size_t)n * 6 + o] = a;
    }
}

static inline int cdiv(long a, long b) { return (int)((a + b - 1) / b); }

static void launch_gemm(const bf16* A, const bf16* Bp, float* C, const float* bias,
                        int M, int K, int Nc, int acc, hipStream_t s) {
    dim3 grid(cdiv(Nc / 16, 8), cdiv(M, 64));
    wmma_gemm_kernel<<<grid, 256, 0, s>>>(A, Bp, C, bias, M, K, Nc, acc);
}

extern "C" void kernel_launch(void* const* d_in, const int* in_sizes, int n_in,
                              void* d_out, int out_size, void* d_ws, size_t ws_size,
                              hipStream_t stream) {
    (void)in_sizes; (void)n_in; (void)out_size; (void)ws_size;
    const float* x      = (const float*)d_in[0];
    const int*   eidx   = (const int*)d_in[1];
    const float* in_W   = (const float*)d_in[2];
    const float* in_b   = (const float*)d_in[3];
    const float* in_lg  = (const float*)d_in[4];
    const float* in_lb  = (const float*)d_in[5];
    const float* gcn_W  = (const float*)d_in[6];
    const float* gcn_b  = (const float*)d_in[7];
    const float* gcn_lg = (const float*)d_in[8];
    const float* gcn_lb = (const float*)d_in[9];
    const float* Wih[4] = { (const float*)d_in[10], (const float*)d_in[13],
                            (const float*)d_in[16], (const float*)d_in[19] };
    const float* Whh[4] = { (const float*)d_in[11], (const float*)d_in[14],
                            (const float*)d_in[17], (const float*)d_in[20] };
    const float* bls[4] = { (const float*)d_in[12], (const float*)d_in[15],
                            (const float*)d_in[18], (const float*)d_in[21] };
    const float* attn_W1 = (const float*)d_in[22];
    const float* attn_b1 = (const float*)d_in[23];
    const float* attn_W2 = (const float*)d_in[24];
    const float* out_W1  = (const float*)d_in[25];
    const float* out_b1  = (const float*)d_in[26];
    const float* out_W2  = (const float*)d_in[27];
    const float* out_b2  = (const float*)d_in[28];
    const float* out_W3  = (const float*)d_in[29];
    const float* out_b3  = (const float*)d_in[30];

    char* base = (char*)d_ws;
    size_t off = 0;
    auto alloc = [&](size_t bytes) { size_t o = off; off += (bytes + 255) & ~(size_t)255; return o; };

    size_t o_di    = alloc((size_t)NN * 4);
    size_t o_h     = alloc((size_t)NN * HH * 4);      // start of reusable contiguous region
    size_t o_hbf   = alloc((size_t)NN * HH * 2);
    size_t o_m     = alloc((size_t)NN * HH * 4);
    size_t o_agg   = alloc((size_t)NN * HH * 4);
    size_t o_gates = alloc((size_t)NN * G4 * 4);
    size_t o_feats = alloc((size_t)TT * NN * HH * 2); // end of reusable region (~138 MB)
    size_t o_c     = alloc((size_t)NN * LHH * 4);
    size_t o_hsbf  = alloc((size_t)NN * LHH * 2);
    size_t o_l1    = alloc((size_t)TT * NN * D2 * 2);
    size_t o_lo    = alloc((size_t)TT * NN * D2 * 2);
    size_t o_sc    = alloc((size_t)TT * NN * 4);
    size_t o_wts   = alloc((size_t)TT * NN * 4);
    size_t o_fin   = alloc((size_t)NN * D2 * 2);
    size_t o_gcnW  = alloc((size_t)3 * HH * HH * 2);
    size_t o_wihT[4], o_whhT[4];
    int Kin[4] = { HH, HH, D2, D2 };
    for (int i = 0; i < 4; ++i) {
        o_wihT[i] = alloc((size_t)Kin[i] * G4 * 2);
        o_whhT[i] = alloc((size_t)LHH * G4 * 2);
    }
    size_t o_aW1 = alloc((size_t)D2 * LHH * 2);
    size_t o_oW1 = alloc((size_t)D2 * HH * 2);
    size_t o_oW2 = alloc((size_t)HH * 64 * 2);

    float* di    = (float*)(base + o_di);
    float* hbuf  = (float*)(base + o_h);
    bf16*  hbf   = (bf16*)(base + o_hbf);
    float* mbuf  = (float*)(base + o_m);
    float* agg   = (float*)(base + o_agg);
    float* gates = (float*)(base + o_gates);
    bf16*  feats = (bf16*)(base + o_feats);
    float* cst   = (float*)(base + o_c);
    bf16*  hsbf  = (bf16*)(base + o_hsbf);
    bf16*  l1bf  = (bf16*)(base + o_l1);
    bf16*  lobf  = (bf16*)(base + o_lo);
    float* sc    = (float*)(base + o_sc);
    float* wts   = (float*)(base + o_wts);
    bf16*  finbf = (bf16*)(base + o_fin);
    bf16*  gcnWp = (bf16*)(base + o_gcnW);
    // aliases into the dead GCN/gates/feats region (safe after their last use)
    float* attn_h = (float*)(base + o_h);   // [T*N,128] f32 (122.9 MB < region)
    float* h1     = (float*)(base + o_h);   // after attention done
    bf16*  h1bf   = (bf16*)(base + o_hbf);
    float* h2     = (float*)(base + o_m);

    // ---- weight prep: pack all GEMM B matrices into WMMA fragment order ----
    for (int i = 0; i < 3; ++i)
        pack_b_kernel<<<cdiv((long)HH * HH, 256), 256, 0, stream>>>(
            gcn_W + (size_t)i * HH * HH, gcnWp + (size_t)i * HH * HH, HH, HH, 0);
    for (int i = 0; i < 4; ++i) {
        pack_b_kernel<<<cdiv((long)Kin[i] * G4, 256), 256, 0, stream>>>(
            Wih[i], (bf16*)(base + o_wihT[i]), Kin[i], G4, 1);
        pack_b_kernel<<<cdiv((long)LHH * G4, 256), 256, 0, stream>>>(
            Whh[i], (bf16*)(base + o_whhT[i]), LHH, G4, 1);
    }
    pack_b_kernel<<<cdiv((long)D2 * LHH, 256), 256, 0, stream>>>(attn_W1, (bf16*)(base + o_aW1), D2, LHH, 0);
    pack_b_kernel<<<cdiv((long)D2 * HH, 256), 256, 0, stream>>>(out_W1, (bf16*)(base + o_oW1), D2, HH, 0);
    pack_b_kernel<<<cdiv((long)HH * 64, 256), 256, 0, stream>>>(out_W2, (bf16*)(base + o_oW2), HH, 64, 0);

    // ---- degree normalization ----
    deg_init_kernel<<<cdiv(NN, 256), 256, 0, stream>>>(di);
    deg_accum_kernel<<<cdiv(EE, 256), 256, 0, stream>>>(eidx, di);
    deg_finalize_kernel<<<cdiv(NN, 256), 256, 0, stream>>>(di);

    // ---- per-timestep: input proj + 3 GCN layers ----
    for (int t = 0; t < TT; ++t) {
        input_proj_kernel<<<cdiv(NN, 8), 256, 0, stream>>>(x, in_W, in_b, in_lg, in_lb, hbuf, hbf, t);
        for (int i = 0; i < 3; ++i) {
            launch_gemm(hbf, gcnWp + (size_t)i * HH * HH, mbuf, nullptr, NN, HH, HH, 0, stream);
            agg_init_kernel<<<cdiv((long)NN * HH, 256), 256, 0, stream>>>(mbuf, di, gcn_b + i * HH, agg);
            edge_scatter_kernel<<<cdiv((long)EE * 32, 256), 256, 0, stream>>>(eidx, mbuf, di, agg);
            bf16* obf = (i == 2) ? (feats + (size_t)t * NN * HH) : hbf;
            gcn_ln_relu_kernel<<<cdiv(NN, 8), 256, 0, stream>>>(agg, gcn_lg + i * HH, gcn_lb + i * HH, hbuf, obf);
        }
    }

    // ---- 2-layer bidirectional LSTM ----
    for (int L = 0; L < 2; ++L) {
        for (int d = 0; d < 2; ++d) {
            int wi = L * 2 + d;
            const bf16* WihP = (const bf16*)(base + o_wihT[wi]);
            const bf16* WhhP = (const bf16*)(base + o_whhT[wi]);
            zero_u32_kernel<<<cdiv((long)NN * LHH, 256), 256, 0, stream>>>((unsigned*)cst, (long)NN * LHH);
            zero_u32_kernel<<<cdiv((long)NN * LHH / 2, 256), 256, 0, stream>>>((unsigned*)hsbf, (long)NN * LHH / 2);
            for (int s = 0; s < TT; ++s) {
                int t = (d == 0) ? s : (TT - 1 - s);
                const bf16* Xt = (L == 0) ? (feats + (size_t)t * NN * HH)
                                          : (l1bf + (size_t)t * NN * D2);
                launch_gemm(Xt, WihP, gates, bls[wi], NN, Kin[wi], G4, 0, stream);
                launch_gemm(hsbf, WhhP, gates, nullptr, NN, LHH, G4, 1, stream);
                bf16* outp = ((L == 0) ? l1bf : lobf) + (size_t)t * NN * D2 + d * LHH;
                lstm_cell_kernel<<<cdiv((long)NN * LHH, 256), 256, 0, stream>>>(gates, cst, hsbf, outp);
            }
        }
    }

    // ---- temporal attention ----
    launch_gemm(lobf, (const bf16*)(base + o_aW1), attn_h, attn_b1, TT * NN, D2, LHH, 0, stream);
    attn_score_kernel<<<cdiv(TT * NN, 256), 256, 0, stream>>>(attn_h, attn_W2, sc, TT * NN);
    softmax_t_kernel<<<cdiv(NN, 256), 256, 0, stream>>>(sc, wts);
    context_kernel<<<cdiv((long)NN * D2, 256), 256, 0, stream>>>(wts, lobf, finbf);

    // ---- output MLP ----
    launch_gemm(finbf, (const bf16*)(base + o_oW1), h1, out_b1, NN, D2, HH, 0, stream);
    gelu_kernel<<<cdiv((long)NN * HH, 256), 256, 0, stream>>>(h1, nullptr, h1bf, (long)NN * HH);
    launch_gemm(h1bf, (const bf16*)(base + o_oW2), h2, out_b2, NN, HH, 64, 0, stream);
    gelu_kernel<<<cdiv((long)NN * 64, 256), 256, 0, stream>>>(h2, h2, nullptr, (long)NN * 64);
    out3_kernel<<<cdiv(NN, 256), 256, 0, stream>>>(h2, out_W3, out_b3, (float*)d_out);
}
